// AuxiliaryHybridRecurrentFFN_69123203662010
// MI455X (gfx1250) — matrix operand order repaired and checked by reading, not compiled, not run on hardware
//
#include <hip/hip_runtime.h>
#include <math.h>

// ---------------------------------------------------------------------------
// AuxiliaryHybridRecurrentFFN for MI455X (gfx1250), bf16 WMMA pipeline.
// B=4, S=2048, H=R=L=2048.  9 GEMMs of 68.7 GFLOP each -> matrix-core bound.
//
// Weights are converted fp32->bf16 AND transposed once (W^T[N][K]) so that
// every GEMM operand tile is a plain row-major 128x64 bf16 tile.  Both A and
// B tiles are then streamed by the Tensor Data Mover (tensor_load_to_lds)
// into LDS double buffers, overlapped with the v_wmma_f32_16x16x32_bf16
// compute via s_wait_tensorcnt.  The GEMM inner loop has no per-thread
// staging instructions at all.
// ---------------------------------------------------------------------------

typedef __attribute__((ext_vector_type(16))) __bf16 v16bf;
typedef __attribute__((ext_vector_type(8)))  __bf16 v8bf;
typedef __attribute__((ext_vector_type(8)))  float  v8f;
typedef unsigned int u32x4 __attribute__((ext_vector_type(4)));
typedef int          i32x4 __attribute__((ext_vector_type(4)));
typedef int          i32x8 __attribute__((ext_vector_type(8)));

static constexpr int kB = 4;
static constexpr int kS = 2048;
static constexpr int kH = 2048;
static constexpr int kR = 2048;
static constexpr int kL = 2048;
static constexpr int kM = kB * kS;        // 8192 rows in every GEMM

static constexpr int BM  = 128;   // block tile M
static constexpr int BN  = 128;   // block tile N
static constexpr int BKT = 64;    // staged K per tile = two wmma K-steps
static constexpr int NTH = 256;   // 8 waves (wave32)

#ifndef __has_builtin
#define __has_builtin(x) 0
#endif
#if __has_builtin(__builtin_amdgcn_tensor_load_to_lds) && \
    __has_builtin(__builtin_amdgcn_s_wait_tensorcnt)
#define USE_TDM 1
#else
#define USE_TDM 0
#endif

// ---------------- scalar helpers ----------------
__device__ __forceinline__ float bf2f(__bf16 b) {
  unsigned short h = __builtin_bit_cast(unsigned short, b);
  unsigned int u = (unsigned int)h << 16;
  return __builtin_bit_cast(float, u);
}
__device__ __forceinline__ __bf16 f2bf(float f) {
  unsigned int u = __builtin_bit_cast(unsigned int, f);
  u += 0x7FFFu + ((u >> 16) & 1u);          // round-to-nearest-even
  unsigned short h = (unsigned short)(u >> 16);
  return __builtin_bit_cast(__bf16, h);
}
__device__ __forceinline__ float sigm_(float x) { return 1.0f / (1.0f + __expf(-x)); }
__device__ __forceinline__ float silu_(float x) { return x * sigm_(x); }

__device__ __forceinline__ v8f wmma_bf16(v16bf a, v16bf b, v8f c) {
  return __builtin_amdgcn_wmma_f32_16x16x32_bf16(false, a, false, b, (short)0, c,
                                                 false, false);
}

// ---------------- TDM: DMA one 128 x 64 bf16 row-major tile into LDS -------
#if USE_TDM
__device__ __forceinline__ void tdm_load_tile(const __bf16* gtile, int ld,
                                              __bf16* lds_dst) {
  unsigned long long ga = (unsigned long long)(const void*)gtile;
  // LDS aperture keeps the byte offset in addr[31:0] (ISA 10.2 aperture calc)
  unsigned lds = (unsigned)(unsigned long long)(const void*)lds_dst;
  u32x4 g0;
  g0[0] = 1u;                                            // count=1, user mode
  g0[1] = lds;                                           // lds_addr
  g0[2] = (unsigned)(ga & 0xFFFFFFFFu);                  // global_addr[31:0]
  g0[3] = (unsigned)((ga >> 32) & 0x01FFFFFFu) | (2u << 30);  // addr[56:32]|type=2
  i32x8 g1;
  const unsigned td1 = 1u << 22;                         // huge row count (no OOB)
  g1[0] = (int)(1u << 16);                               // data_size=1 -> 2 bytes
  g1[1] = (int)(((unsigned)ld & 0xFFFFu) << 16);         // tensor_dim0 lo16
  g1[2] = (int)((((unsigned)ld >> 16) & 0xFFFFu) |
                ((td1 & 0xFFFFu) << 16));                // dim0 hi | tensor_dim1 lo
  g1[3] = (int)(((td1 >> 16) & 0xFFFFu) |
                ((unsigned)BKT << 16));                  // dim1 hi | tile_dim0=64
  g1[4] = (int)(unsigned)BM;                             // tile_dim1=128 | dim2=0
  g1[5] = (int)(unsigned)ld;                             // tensor_dim0_stride lo32
  g1[6] = 0;
  g1[7] = 0;
  i32x4 z = {0, 0, 0, 0};
#if __clang_major__ >= 23
  i32x8 z8 = {0, 0, 0, 0, 0, 0, 0, 0};
  __builtin_amdgcn_tensor_load_to_lds(g0, g1, z, z, z8, 0);
#else
  __builtin_amdgcn_tensor_load_to_lds(g0, g1, z, z, 0);
#endif
}
#endif

// Fallback cooperative load: 128x64 row-major tile, 1024 8-elem chunks.
__device__ __forceinline__ void coop_load_tile64(const __bf16* __restrict__ g, int ld,
                                                 __bf16* s, int tid) {
#pragma unroll
  for (int i = 0; i < 4; ++i) {
    int c   = tid + i * NTH;
    int row = c >> 3;            // 8 chunks per row of 64
    int col = (c & 7) << 3;
    *(v8bf*)(s + row * BKT + col) = *(const v8bf*)(g + (size_t)row * ld + col);
  }
}

// ---------------- fragment loads (per ISA 7.12.2 layouts) ----------------
// A 16x32 bf16: lanes 0-15 rows, K {0..7,16..23}; lanes 16-31: {8..15,24..31}
__device__ __forceinline__ v16bf load_a_frag64(const __bf16* As, int am, int ks,
                                               int lane) {
  int hf = lane >> 4, m = lane & 15;
  const __bf16* row = As + (am + m) * BKT + ks * 32;
  v8bf lo = *(const v8bf*)(row + hf * 8);
  v8bf hi = *(const v8bf*)(row + 16 + hf * 8);
  v16bf a;
#pragma unroll
  for (int e = 0; e < 8; ++e) { a[e] = lo[e]; a[e + 8] = hi[e]; }
  return a;
}
// B 32x16 bf16: lanes 0-15 cols, K {0..7,16..23}; lanes 16-31: {8..15,24..31}
// Bs holds W^T rows: Bs[n][k], so the per-lane K-run is contiguous.
__device__ __forceinline__ v16bf load_b_frag64(const __bf16* Bs, int bn, int ks,
                                               int lane) {
  int hf = lane >> 4, n = lane & 15;
  const __bf16* row = Bs + (bn + n) * BKT + ks * 32;
  v8bf lo = *(const v8bf*)(row + hf * 8);
  v8bf hi = *(const v8bf*)(row + 16 + hf * 8);
  v16bf b;
#pragma unroll
  for (int e = 0; e < 8; ++e) { b[e] = lo[e]; b[e + 8] = hi[e]; }
  return b;
}

// ---------------- fp32 -> bf16 convert (straight) ----------------
__global__ __launch_bounds__(NTH) void k_cvt_bf16(const float* __restrict__ src,
                                                  __bf16* __restrict__ dst, int n) {
  int i = blockIdx.x * blockDim.x + threadIdx.x;
  int stride = gridDim.x * blockDim.x;
  for (; i < n; i += stride) dst[i] = f2bf(src[i]);
}

// ---------------- fp32 [K][N] -> bf16 transposed [N][K], 64x64 LDS tiles ---
__global__ __launch_bounds__(NTH) void k_cvt_bf16_t(const float* __restrict__ src,
                                                    __bf16* __restrict__ dst,
                                                    int K, int N) {
  __shared__ __bf16 tile[64][65];            // +1 pad: conflict-free transpose
  const int k0 = blockIdx.x * 64, n0 = blockIdx.y * 64;
  const int t = threadIdx.x;
#pragma unroll
  for (int i = 0; i < 16; ++i) {
    int idx = t + NTH * i;
    int r = idx >> 6, c = idx & 63;
    tile[r][c] = f2bf(src[(size_t)(k0 + r) * N + (n0 + c)]);
  }
  __syncthreads();
#pragma unroll
  for (int i = 0; i < 16; ++i) {
    int idx = t + NTH * i;
    int r = idx >> 6, c = idx & 63;
    dst[(size_t)(n0 + r) * K + (k0 + c)] = tile[c][r];
  }
}

// ---------------- single-B GEMM: C(bf16) = ep(A @ B^T^T) -------------------
// A: [M][K] bf16;  Bt: W^T [N][K] bf16;  EP: 0 = tanh, 1 = identity
template <int EP>
__global__ __launch_bounds__(NTH) void k_gemm_uni(const __bf16* __restrict__ A,
                                                  const __bf16* __restrict__ Bt,
                                                  __bf16* __restrict__ C,
                                                  int K, int lda, int ldbt, int ldc) {
  __shared__ __align__(16) __bf16 As[2 * BM * BKT];
  __shared__ __align__(16) __bf16 Bs[2 * BN * BKT];
  const int tid = threadIdx.x, lane = tid & 31, wave = tid >> 5;
  const int wm = wave & 3, wn = wave >> 2;            // 4x2 wave grid
  const int m0 = blockIdx.y * BM, n0 = blockIdx.x * BN;
  const int nk = K / BKT;
  const __bf16* Abase = A + (size_t)m0 * lda;
  const __bf16* Bbase = Bt + (size_t)n0 * ldbt;
#if USE_TDM
  if (wave == 0) {
    tdm_load_tile(Abase, lda, As);
    tdm_load_tile(Bbase, ldbt, Bs);
  }
#endif
  v8f acc[2][4] = {};
  for (int kt = 0; kt < nk; ++kt) {
    const int cur = (kt & 1), nxt = (kt + 1) & 1;
#if USE_TDM
    if (wave == 0) {
      __builtin_amdgcn_s_wait_tensorcnt(0);       // tiles kt have landed
      if (kt + 1 < nk) {                          // stream next pair during compute
        tdm_load_tile(Abase + (size_t)(kt + 1) * BKT, lda, As + nxt * (BM * BKT));
        tdm_load_tile(Bbase + (size_t)(kt + 1) * BKT, ldbt, Bs + nxt * (BN * BKT));
      }
    }
#else
    coop_load_tile64(Abase + (size_t)kt * BKT, lda, As + cur * (BM * BKT), tid);
    coop_load_tile64(Bbase + (size_t)kt * BKT, ldbt, Bs + cur * (BN * BKT), tid);
    (void)nxt;
#endif
    __syncthreads();
    const __bf16* Ab = As + cur * (BM * BKT);
    const __bf16* Bb = Bs + cur * (BN * BKT);
#pragma unroll
    for (int ks = 0; ks < 2; ++ks) {
      v16bf af0 = load_a_frag64(Ab, wm * 32, ks, lane);
      v16bf af1 = load_a_frag64(Ab, wm * 32 + 16, ks, lane);
#pragma unroll
      for (int in = 0; in < 4; ++in) {
        v16bf bf_ = load_b_frag64(Bb, wn * 64 + in * 16, ks, lane);
        acc[0][in] = wmma_bf16(af0, bf_, acc[0][in]);
        acc[1][in] = wmma_bf16(af1, bf_, acc[1][in]);
      }
    }
    __syncthreads();
  }
  const int hf = lane >> 4, nn = lane & 15;
#pragma unroll
  for (int im = 0; im < 2; ++im)
#pragma unroll
    for (int in = 0; in < 4; ++in)
#pragma unroll
      for (int v = 0; v < 8; ++v) {
        int row = m0 + wm * 32 + im * 16 + v + hf * 8;
        int col = n0 + wn * 64 + in * 16 + nn;
        float val = acc[im][in][v];
        if (EP == 0) val = tanhf(val);
        C[(size_t)row * ldc + col] = f2bf(val);
      }
}

// ---------------- dual-B GEMM: C = ep(A@B0, A@B1) ----------------
// EP: 0 -> sigmoid(p0)*silu(p1)     1 -> p0*silu(p1)
template <int EP>
__global__ __launch_bounds__(NTH) void k_gemm_dual(const __bf16* __restrict__ A,
                                                   const __bf16* __restrict__ B0t,
                                                   const __bf16* __restrict__ B1t,
                                                   __bf16* __restrict__ C,
                                                   int K, int lda, int ldbt, int ldc) {
  __shared__ __align__(16) __bf16 As[2 * BM * BKT];
  __shared__ __align__(16) __bf16 Bs0[2 * BN * BKT];
  __shared__ __align__(16) __bf16 Bs1[2 * BN * BKT];
  const int tid = threadIdx.x, lane = tid & 31, wave = tid >> 5;
  const int wm = wave & 3, wn = wave >> 2;
  const int m0 = blockIdx.y * BM, n0 = blockIdx.x * BN;
  const int nk = K / BKT;
  const __bf16* Abase = A + (size_t)m0 * lda;
  const __bf16* B0base = B0t + (size_t)n0 * ldbt;
  const __bf16* B1base = B1t + (size_t)n0 * ldbt;
#if USE_TDM
  if (wave == 0) {
    tdm_load_tile(Abase, lda, As);
    tdm_load_tile(B0base, ldbt, Bs0);
    tdm_load_tile(B1base, ldbt, Bs1);
  }
#endif
  v8f acc0[2][4] = {}, acc1[2][4] = {};
  for (int kt = 0; kt < nk; ++kt) {
    const int cur = (kt & 1), nxt = (kt + 1) & 1;
#if USE_TDM
    if (wave == 0) {
      __builtin_amdgcn_s_wait_tensorcnt(0);
      if (kt + 1 < nk) {
        tdm_load_tile(Abase + (size_t)(kt + 1) * BKT, lda, As + nxt * (BM * BKT));
        tdm_load_tile(B0base + (size_t)(kt + 1) * BKT, ldbt, Bs0 + nxt * (BN * BKT));
        tdm_load_tile(B1base + (size_t)(kt + 1) * BKT, ldbt, Bs1 + nxt * (BN * BKT));
      }
    }
#else
    coop_load_tile64(Abase + (size_t)kt * BKT, lda, As + cur * (BM * BKT), tid);
    coop_load_tile64(B0base + (size_t)kt * BKT, ldbt, Bs0 + cur * (BN * BKT), tid);
    coop_load_tile64(B1base + (size_t)kt * BKT, ldbt, Bs1 + cur * (BN * BKT), tid);
    (void)nxt;
#endif
    __syncthreads();
    const __bf16* Ab = As + cur * (BM * BKT);
    const __bf16* Bb0 = Bs0 + cur * (BN * BKT);
    const __bf16* Bb1 = Bs1 + cur * (BN * BKT);
#pragma unroll
    for (int ks = 0; ks < 2; ++ks) {
      v16bf af0 = load_a_frag64(Ab, wm * 32, ks, lane);
      v16bf af1 = load_a_frag64(Ab, wm * 32 + 16, ks, lane);
#pragma unroll
      for (int in = 0; in < 4; ++in) {
        v16bf b0 = load_b_frag64(Bb0, wn * 64 + in * 16, ks, lane);
        acc0[0][in] = wmma_bf16(af0, b0, acc0[0][in]);
        acc0[1][in] = wmma_bf16(af1, b0, acc0[1][in]);
        v16bf b1 = load_b_frag64(Bb1, wn * 64 + in * 16, ks, lane);
        acc1[0][in] = wmma_bf16(af0, b1, acc1[0][in]);
        acc1[1][in] = wmma_bf16(af1, b1, acc1[1][in]);
      }
    }
    __syncthreads();
  }
  const int hf = lane >> 4, nn = lane & 15;
#pragma unroll
  for (int im = 0; im < 2; ++im)
#pragma unroll
    for (int in = 0; in < 4; ++in)
#pragma unroll
      for (int v = 0; v < 8; ++v) {
        int row = m0 + wm * 32 + im * 16 + v + hf * 8;
        int col = n0 + wn * 64 + in * 16 + nn;
        float p0 = acc0[im][in][v], p1 = acc1[im][in][v];
        float val = (EP == 0) ? sigm_(p0) * silu_(p1) : p0 * silu_(p1);
        C[(size_t)row * ldc + col] = f2bf(val);
      }
}

// ---------------- dual-K output GEMM: out = A1@B1 + A2@B2 (fp32 store) -----
__global__ __launch_bounds__(NTH) void k_gemm_out(const __bf16* __restrict__ A1,
                                                  const __bf16* __restrict__ B1t,
                                                  const __bf16* __restrict__ A2,
                                                  const __bf16* __restrict__ B2t,
                                                  float* __restrict__ C,
                                                  int K, int lda, int ldbt, int ldc) {
  __shared__ __align__(16) __bf16 As[2 * BM * BKT];
  __shared__ __align__(16) __bf16 Bs[2 * BN * BKT];
  const int tid = threadIdx.x, lane = tid & 31, wave = tid >> 5;
  const int wm = wave & 3, wn = wave >> 2;
  const int m0 = blockIdx.y * BM, n0 = blockIdx.x * BN;
  const int nk = K / BKT;
  v8f acc[2][4] = {};
  for (int ph = 0; ph < 2; ++ph) {
    const __bf16* Abase = (ph ? A2 : A1) + (size_t)m0 * lda;
    const __bf16* Bbase = (ph ? B2t : B1t) + (size_t)n0 * ldbt;
#if USE_TDM
    if (wave == 0) {
      tdm_load_tile(Abase, lda, As);
      tdm_load_tile(Bbase, ldbt, Bs);
    }
#endif
    for (int kt = 0; kt < nk; ++kt) {
      const int cur = (kt & 1), nxt = (kt + 1) & 1;
#if USE_TDM
      if (wave == 0) {
        __builtin_amdgcn_s_wait_tensorcnt(0);
        if (kt + 1 < nk) {
          tdm_load_tile(Abase + (size_t)(kt + 1) * BKT, lda, As + nxt * (BM * BKT));
          tdm_load_tile(Bbase + (size_t)(kt + 1) * BKT, ldbt, Bs + nxt * (BN * BKT));
        }
      }
#else
      coop_load_tile64(Abase + (size_t)kt * BKT, lda, As + cur * (BM * BKT), tid);
      coop_load_tile64(Bbase + (size_t)kt * BKT, ldbt, Bs + cur * (BN * BKT), tid);
      (void)nxt;
#endif
      __syncthreads();
      const __bf16* Ab = As + cur * (BM * BKT);
      const __bf16* Bb = Bs + cur * (BN * BKT);
#pragma unroll
      for (int ks = 0; ks < 2; ++ks) {
        v16bf af0 = load_a_frag64(Ab, wm * 32, ks, lane);
        v16bf af1 = load_a_frag64(Ab, wm * 32 + 16, ks, lane);
#pragma unroll
        for (int in = 0; in < 4; ++in) {
          v16bf bf_ = load_b_frag64(Bb, wn * 64 + in * 16, ks, lane);
          acc[0][in] = wmma_bf16(af0, bf_, acc[0][in]);
          acc[1][in] = wmma_bf16(af1, bf_, acc[1][in]);
        }
      }
      __syncthreads();
    }
  }
  const int hf = lane >> 4, nn = lane & 15;
#pragma unroll
  for (int im = 0; im < 2; ++im)
#pragma unroll
    for (int in = 0; in < 4; ++in)
#pragma unroll
      for (int v = 0; v < 8; ++v) {
        int row = m0 + wm * 32 + im * 16 + v + hf * 8;
        int col = n0 + wn * 64 + in * 16 + nn;
        C[(size_t)row * ldc + col] = acc[im][in][v];
      }
}

// ---------------- aux GEMM + deterministic squared-error reduction ---------
__global__ __launch_bounds__(NTH) void k_gemm_aux(const __bf16* __restrict__ A,
                                                  const __bf16* __restrict__ Bwt,
                                                  const float* __restrict__ xin,
                                                  float* __restrict__ partial,
                                                  int K, int lda, int ldbt, int N,
                                                  int Sseq) {
  __shared__ __align__(16) __bf16 As[2 * BM * BKT];
  __shared__ __align__(16) __bf16 Bs[2 * BN * BKT];
  __shared__ float red[NTH];
  const int tid = threadIdx.x, lane = tid & 31, wave = tid >> 5;
  const int wm = wave & 3, wn = wave >> 2;
  const int m0 = blockIdx.y * BM, n0 = blockIdx.x * BN;
  const int nk = K / BKT;
  const __bf16* Abase = A + (size_t)m0 * lda;
  const __bf16* Bbase = Bwt + (size_t)n0 * ldbt;
#if USE_TDM
  if (wave == 0) {
    tdm_load_tile(Abase, lda, As);
    tdm_load_tile(Bbase, ldbt, Bs);
  }
#endif
  v8f acc[2][4] = {};
  for (int kt = 0; kt < nk; ++kt) {
    const int cur = (kt & 1), nxt = (kt + 1) & 1;
#if USE_TDM
    if (wave == 0) {
      __builtin_amdgcn_s_wait_tensorcnt(0);
      if (kt + 1 < nk) {
        tdm_load_tile(Abase + (size_t)(kt + 1) * BKT, lda, As + nxt * (BM * BKT));
        tdm_load_tile(Bbase + (size_t)(kt + 1) * BKT, ldbt, Bs + nxt * (BN * BKT));
      }
    }
#else
    coop_load_tile64(Abase + (size_t)kt * BKT, lda, As + cur * (BM * BKT), tid);
    coop_load_tile64(Bbase + (size_t)kt * BKT, ldbt, Bs + cur * (BN * BKT), tid);
    (void)nxt;
#endif
    __syncthreads();
    const __bf16* Ab = As + cur * (BM * BKT);
    const __bf16* Bb = Bs + cur * (BN * BKT);
#pragma unroll
    for (int ks = 0; ks < 2; ++ks) {
      v16bf af0 = load_a_frag64(Ab, wm * 32, ks, lane);
      v16bf af1 = load_a_frag64(Ab, wm * 32 + 16, ks, lane);
#pragma unroll
      for (int in = 0; in < 4; ++in) {
        v16bf bf_ = load_b_frag64(Bb, wn * 64 + in * 16, ks, lane);
        acc[0][in] = wmma_bf16(af0, bf_, acc[0][in]);
        acc[1][in] = wmma_bf16(af1, bf_, acc[1][in]);
      }
    }
    __syncthreads();
  }
  const int hf = lane >> 4, nn = lane & 15;
  float local = 0.0f;
#pragma unroll
  for (int im = 0; im < 2; ++im)
#pragma unroll
    for (int in = 0; in < 4; ++in)
#pragma unroll
      for (int v = 0; v < 8; ++v) {
        int row = m0 + wm * 32 + im * 16 + v + hf * 8;
        int col = n0 + wn * 64 + in * 16 + nn;
        if ((row % Sseq) != Sseq - 1) {            // states[:, :-1] vs x[:, 1:]
          float d = acc[im][in][v] - xin[(size_t)(row + 1) * N + col];
          local += d * d;
        }
      }
  red[tid] = local;
  __syncthreads();
  for (int s = NTH / 2; s > 0; s >>= 1) {
    if (tid < s) red[tid] += red[tid + s];
    __syncthreads();
  }
  if (tid == 0) partial[blockIdx.y * gridDim.x + blockIdx.x] = red[0];
}

__global__ __launch_bounds__(NTH) void k_aux_finalize(const float* __restrict__ partial,
                                                      int n, float* __restrict__ dst,
                                                      float inv) {
  __shared__ float red[NTH];
  float s = 0.0f;
  for (int i = threadIdx.x; i < n; i += NTH) s += partial[i];
  red[threadIdx.x] = s;
  __syncthreads();
  for (int t = NTH / 2; t > 0; t >>= 1) {
    if (threadIdx.x < t) red[threadIdx.x] += red[threadIdx.x + t];
    __syncthreads();
  }
  if (threadIdx.x == 0) *dst = red[0] * inv;
}

// ---------------- sequential scan over time, parallel over (b, r) ----------
__global__ __launch_bounds__(NTH) void k_scan(const __bf16* __restrict__ fb,
                                              const __bf16* __restrict__ gb,
                                              const __bf16* __restrict__ qb,
                                              const float* __restrict__ init,
                                              __bf16* __restrict__ rb,
                                              __bf16* __restrict__ ab) {
  int ch = blockIdx.x * blockDim.x + threadIdx.x;   // 0 .. B*R-1
  if (ch >= kB * kR) return;
  int b = ch / kR, r = ch % kR;
  float st = init[r];
  size_t idx = (size_t)b * kS * kR + r;
  for (int t = 0; t < kS; ++t) {
    float f = bf2f(fb[idx]);
    float g = bf2f(gb[idx]);
    st = f * st + g;
    float q = bf2f(qb[idx]);
    rb[idx] = f2bf(silu_(q * st));   // readout = silu(q * state)
    ab[idx] = f2bf(silu_(st));       // aux input = silu(state)
    idx += (size_t)kR;
  }
}

// ---------------------------------------------------------------------------
extern "C" void kernel_launch(void* const* d_in, const int* in_sizes, int n_in,
                              void* d_out, int out_size, void* d_ws, size_t ws_size,
                              hipStream_t stream) {
  const float* x      = (const float*)d_in[0];
  const float* w_f    = (const float*)d_in[1];
  const float* w_i    = (const float*)d_in[2];
  const float* w_v    = (const float*)d_in[3];
  const float* w_q    = (const float*)d_in[4];
  const float* w_ro   = (const float*)d_in[5];
  const float* w_aux  = (const float*)d_in[6];
  const float* w_up   = (const float*)d_in[7];
  const float* w_gate = (const float*)d_in[8];
  const float* w_down = (const float*)d_in[9];
  const float* init_s = (const float*)d_in[10];
  float* out = (float*)d_out;

  const size_t XB = (size_t)kB * kS * kH;   // 16,777,216
  const size_t WE = (size_t)kH * kR;        //  4,194,304

  // workspace layout (bf16 regions then fp32 partials); ~310 MB total
  __bf16* p = (__bf16*)d_ws;
  __bf16* xb = p;  p += XB;
  __bf16* wfb = p; p += WE;  __bf16* wib = p;   p += WE;   // transposed weights
  __bf16* wvb = p; p += WE;  __bf16* wqb = p;   p += WE;
  __bf16* wrob = p; p += WE; __bf16* wauxb = p; p += WE;
  __bf16* wupb = p; p += WE; __bf16* wgb = p;   p += WE;
  __bf16* wdb = p; p += WE;
  __bf16* fb = p; p += XB;   __bf16* gb = p; p += XB;
  __bf16* qb = p; p += XB;   __bf16* rb = p; p += XB;
  __bf16* ab = p; p += XB;   __bf16* hb = p; p += XB;
  float* partials = (float*)p;              // 1024 floats

  const dim3 blk(NTH);
  // 1) convert x (straight) and weights (transposing) to bf16
  k_cvt_bf16<<<dim3(2048), blk, 0, stream>>>(x, xb, (int)XB);
  const dim3 tg(kH / 64, kR / 64);          // (32, 32)
  k_cvt_bf16_t<<<tg, blk, 0, stream>>>(w_f, wfb, kH, kR);
  k_cvt_bf16_t<<<tg, blk, 0, stream>>>(w_i, wib, kH, kR);
  k_cvt_bf16_t<<<tg, blk, 0, stream>>>(w_v, wvb, kH, kR);
  k_cvt_bf16_t<<<tg, blk, 0, stream>>>(w_q, wqb, kH, kR);
  k_cvt_bf16_t<<<tg, blk, 0, stream>>>(w_ro, wrob, kR, kH);
  k_cvt_bf16_t<<<tg, blk, 0, stream>>>(w_aux, wauxb, kR, kH);
  k_cvt_bf16_t<<<tg, blk, 0, stream>>>(w_up, wupb, kH, kL);
  k_cvt_bf16_t<<<tg, blk, 0, stream>>>(w_gate, wgb, kH, kL);
  k_cvt_bf16_t<<<tg, blk, 0, stream>>>(w_down, wdb, kL, kH);

  const dim3 gg(kR / BN, kM / BM);  // (16, 64)
  // 2) projections (weights passed transposed: W^T[N][K], ldbt = K)
  k_gemm_uni<0><<<gg, blk, 0, stream>>>(xb, wfb, fb, kH, kH, kH, kR);          // f = tanh
  k_gemm_dual<0><<<gg, blk, 0, stream>>>(xb, wib, wvb, gb, kH, kH, kH, kR);    // g = sig*silu
  k_gemm_uni<1><<<gg, blk, 0, stream>>>(xb, wqb, qb, kH, kH, kH, kR);          // q
  k_gemm_dual<1><<<gg, blk, 0, stream>>>(xb, wupb, wgb, hb, kH, kH, kH, kL);   // h = up*silu(gate)
  // 3) recurrence + readout/aux activations
  k_scan<<<dim3((kB * kR) / NTH), blk, 0, stream>>>(fb, gb, qb, init_s, rb, ab);
  // 4) out = readout@w_ro + h@w_down
  k_gemm_out<<<gg, blk, 0, stream>>>(rb, wrob, hb, wdb, out, kR, kR, kR, kH);
  // 5) aux loss (deterministic two-pass reduction)
  k_gemm_aux<<<gg, blk, 0, stream>>>(ab, wauxb, x, partials, kR, kR, kR, kH, kS);
  k_aux_finalize<<<1, blk, 0, stream>>>(partials, (int)(gg.x * gg.y), out + XB,
                                        1.0f / ((float)kB * (float)(kS - 1) * (float)kH));
}
